// TransformerBlock_46591805227456
// MI455X (gfx1250) — matrix operand order repaired and checked by reading, not compiled, not run on hardware
//
#include <hip/hip_runtime.h>
#include <hip/hip_bf16.h>
#include <cmath>

typedef __bf16 bf16_t;
typedef __attribute__((ext_vector_type(16))) __bf16 v16bf;
typedef __attribute__((ext_vector_type(8)))  __bf16 v8bf;
typedef __attribute__((ext_vector_type(8)))  float  v8f;

#define BB 2
#define SS 2048
#define DD 2048
#define HH 16
#define HDIM 128
#define FFD 8192

static __device__ __forceinline__ v8f wmma_bf16(v16bf a, v16bf b, v8f c) {
  // emits v_wmma_f32_16x16x32_bf16
  return __builtin_amdgcn_wmma_f32_16x16x32_bf16(false, a, false, b, (short)0, c, false, false);
}
static __device__ __forceinline__ v16bf cat8(v8bf lo, v8bf hi) {
  return __builtin_shufflevector(lo, hi, 0,1,2,3,4,5,6,7,8,9,10,11,12,13,14,15);
}
static __device__ __forceinline__ v8f vzero8() {
  v8f z;
#pragma unroll
  for (int i = 0; i < 8; ++i) z[i] = 0.0f;
  return z;
}

// CDNA5 async memory->LDS copy (ASYNCcnt-tracked). vdst = LDS byte address,
// vaddr = 64-bit global address (GV mode). 16B per lane per instruction.
static __device__ __forceinline__ void async_ld_b128(unsigned lds_addr, const void* gaddr) {
  asm volatile("global_load_async_to_lds_b128 %0, %1, off"
               :: "v"(lds_addr), "v"(gaddr) : "memory");
}
static __device__ __forceinline__ void wait_async4() {
  asm volatile("s_wait_asynccnt 0x4" ::: "memory");
}
static __device__ __forceinline__ void wait_async0() {
  asm volatile("s_wait_asynccnt 0x0" ::: "memory");
}

// ---------------------------------------------------------------- fp32 -> bf16
__global__ void cvt_f32_bf16(const float* __restrict__ src, bf16_t* __restrict__ dst, int n) {
  int i = (blockIdx.x * blockDim.x + threadIdx.x) * 4;
  if (i + 3 < n) {
    float4 v = *(const float4*)(src + i);
    dst[i + 0] = (bf16_t)v.x;
    dst[i + 1] = (bf16_t)v.y;
    dst[i + 2] = (bf16_t)v.z;
    dst[i + 3] = (bf16_t)v.w;
  } else {
    for (int k = i; k < n; ++k) dst[k] = (bf16_t)src[k];
  }
}

// ---------------------------------------------------------------- RMSNorm -> bf16
__global__ void rmsnorm_bf16(const float* __restrict__ x, const float* __restrict__ g,
                             bf16_t* __restrict__ out) {
  __shared__ float red[256];
  const int row = blockIdx.x;
  const float* xr = x + (size_t)row * DD;
  float s = 0.0f;
  for (int c = threadIdx.x; c < DD; c += 256) { float v = xr[c]; s += v * v; }
  red[threadIdx.x] = s;
  __syncthreads();
  for (int st = 128; st > 0; st >>= 1) {
    if ((int)threadIdx.x < st) red[threadIdx.x] += red[threadIdx.x + st];
    __syncthreads();
  }
  const float inv = rsqrtf(red[0] * (1.0f / (float)DD) + 1e-5f);
  for (int c = threadIdx.x; c < DD; c += 256)
    out[(size_t)row * DD + c] = (bf16_t)(xr[c] * inv * g[c]);
}

// ---------------------------------------------------------------- WMMA GEMM
// C[M,N] = A[M,K](bf16,row-major) * B[N,K](bf16,row-major)^T, f32 accumulate.
// Double-buffered LDS tiles filled with global_load_async_to_lds_b128.
// MODE 0: store f32, MODE 1: store bf16, MODE 2: store f32 = resid + C.
template <int MODE>
__global__ __launch_bounds__(256) void gemm_bf16(
    const bf16_t* __restrict__ A, const bf16_t* __restrict__ Bw,
    void* __restrict__ Cout, const float* __restrict__ resid,
    int M, int N, int K) {
  // per buffer: A tile 128x32 (8KB) then B tile 128x32 (8KB)
  __shared__ bf16_t lds[2][2 * 128 * 32];

  const int tid  = threadIdx.x;
  const int lane = tid & 31;
  const int wave = tid >> 5;
  const int wm = wave >> 2;        // 0..1
  const int wn = wave & 3;         // 0..3
  const int m16 = lane & 15;
  const int half = lane >> 4;
  const int blockM = blockIdx.y * 128;
  const int blockN = blockIdx.x * 128;

  v8f acc[4][2];
#pragma unroll
  for (int i = 0; i < 4; ++i)
#pragma unroll
    for (int j = 0; j < 2; ++j) acc[i][j] = vzero8();

  const int r = tid >> 2;          // 0..63
  const int c = (tid & 3) << 3;    // 0,8,16,24
  const bf16_t* Ag0 = A  + (size_t)(blockM + r) * K + c;
  const bf16_t* Ag1 = Ag0 + (size_t)64 * K;
  const bf16_t* Bg0 = Bw + (size_t)(blockN + r) * K + c;
  const bf16_t* Bg1 = Bg0 + (size_t)64 * K;

  // LDS byte addresses (generic shared address truncates to LDS offset)
  const unsigned ldsbase = (unsigned)(size_t)(&lds[0][0]);
  const unsigned bufstride = 2u * 128u * 32u * 2u;          // 16KB
  const unsigned offA0 = (unsigned)((r * 32 + c) * 2);
  const unsigned offA1 = (unsigned)(((r + 64) * 32 + c) * 2);
  const unsigned offB0 = offA0 + 128u * 32u * 2u;
  const unsigned offB1 = offA1 + 128u * 32u * 2u;

  const int nt = K >> 5;
  // prologue: async-fill buffer 0 with K-tile 0
  {
    unsigned bb = ldsbase;
    async_ld_b128(bb + offA0, Ag0);
    async_ld_b128(bb + offA1, Ag1);
    async_ld_b128(bb + offB0, Bg0);
    async_ld_b128(bb + offB1, Bg1);
  }

  for (int i = 0; i < nt; ++i) {
    const int cur = i & 1;
    if (i + 1 < nt) {
      const int kt = (i + 1) << 5;
      unsigned bb = ldsbase + (unsigned)(cur ^ 1) * bufstride;
      async_ld_b128(bb + offA0, Ag0 + kt);
      async_ld_b128(bb + offA1, Ag1 + kt);
      async_ld_b128(bb + offB0, Bg0 + kt);
      async_ld_b128(bb + offB1, Bg1 + kt);
      wait_async4();   // tile i complete; tile i+1 may remain in flight
    } else {
      wait_async0();
    }
    __syncthreads();   // all waves' tile-i data visible

    const bf16_t* la = &lds[cur][0];
    const bf16_t* lb = &lds[cur][128 * 32];

    v16bf af[4], bfr[2];
    const int koff = half * 8;
#pragma unroll
    for (int ii = 0; ii < 4; ++ii) {
      int row = wm * 64 + ii * 16 + m16;
      af[ii] = cat8(*(const v8bf*)&la[row * 32 + koff],
                    *(const v8bf*)&la[row * 32 + koff + 16]);
    }
    const int k0 = half * 8;
#pragma unroll
    for (int j = 0; j < 2; ++j) {
      int row = wn * 32 + j * 16 + m16;
      bfr[j] = cat8(*(const v8bf*)&lb[row * 32 + k0],
                    *(const v8bf*)&lb[row * 32 + k0 + 16]);
    }
#pragma unroll
    for (int ii = 0; ii < 4; ++ii)
#pragma unroll
      for (int j = 0; j < 2; ++j) acc[ii][j] = wmma_bf16(af[ii], bfr[j], acc[ii][j]);

    __syncthreads();   // done reading buf[cur]; safe for async writes of tile i+2
  }

#pragma unroll
  for (int i = 0; i < 4; ++i)
#pragma unroll
    for (int j = 0; j < 2; ++j)
#pragma unroll
      for (int e = 0; e < 8; ++e) {
        int row = blockM + wm * 64 + i * 16 + e + 8 * half;
        int col = blockN + wn * 32 + j * 16 + m16;
        size_t idx = (size_t)row * N + col;
        float v = acc[i][j][e];
        if (MODE == 0)      ((float*)Cout)[idx]  = v;
        else if (MODE == 1) ((bf16_t*)Cout)[idx] = (bf16_t)v;
        else                ((float*)Cout)[idx]  = resid[idx] + v;
      }
}

// ---------------------------------------------------------------- RoPE + split/transpose
// qkv f32 [B*S, 3D] -> q(scaled,roped) [B,H,S,HD] bf16, k(roped) [B,H,S,HD] bf16,
// v transposed [B,H,HD,S] bf16.  One thread per (b,s,h, pair i).
__global__ void rope_split(const float* __restrict__ qkv, bf16_t* __restrict__ qo,
                           bf16_t* __restrict__ ko, bf16_t* __restrict__ vo) {
  int t = blockIdx.x * 256 + threadIdx.x;       // < B*S*H*64 = 2^22
  int i = t & 63;
  int h = (t >> 6) & 15;
  int s = (t >> 10) & 2047;
  int b = t >> 21;
  size_t rowbase = (size_t)(b * SS + s) * (3 * DD);
  int col = h * HDIM + 2 * i;
  float q0 = qkv[rowbase + col],            q1 = qkv[rowbase + col + 1];
  float k0 = qkv[rowbase + DD + col],       k1 = qkv[rowbase + DD + col + 1];
  float v0 = qkv[rowbase + 2 * DD + col],   v1 = qkv[rowbase + 2 * DD + col + 1];
  float ang = (float)s * __powf(10000.0f, -(float)i * (1.0f / 64.0f));
  float sn, cs;
  __sincosf(ang, &sn, &cs);
  const float qscale = 0.08838834764831845f;  // 1/sqrt(128)
  size_t qkb = ((size_t)(b * HH + h) * SS + s) * HDIM + 2 * i;
  qo[qkb]     = (bf16_t)((q0 * cs - q1 * sn) * qscale);
  qo[qkb + 1] = (bf16_t)((q0 * sn + q1 * cs) * qscale);
  ko[qkb]     = (bf16_t)(k0 * cs - k1 * sn);
  ko[qkb + 1] = (bf16_t)(k0 * sn + k1 * cs);
  size_t vb = ((size_t)(b * HH + h) * HDIM + 2 * i) * SS + s;
  vo[vb]      = (bf16_t)v0;
  vo[vb + SS] = (bf16_t)v1;
}

// ---------------------------------------------------------------- Flash attention
// one wave per (b, h, 16-query tile); causal, online softmax in f32.
__global__ __launch_bounds__(32) void attn_kernel(const bf16_t* __restrict__ qb,
                                                  const bf16_t* __restrict__ kb,
                                                  const bf16_t* __restrict__ vt,
                                                  bf16_t* __restrict__ ctx) {
  __shared__ bf16_t pls[16 * 32];
  const int bid = blockIdx.x;
  const int qt = bid & ((SS / 16) - 1);
  const int h  = (bid / (SS / 16)) & (HH - 1);
  const int b  = bid / ((SS / 16) * HH);
  const int lane = threadIdx.x;
  const int m = lane & 15, half = lane >> 4;
  const int q0 = qt * 16;
  const size_t bh = (size_t)(b * HH + h);

  // load Q fragments (A-operand layout), 16x128 tile
  v16bf qf[4];
  const bf16_t* qrow = qb + (bh * SS + q0 + m) * HDIM;
#pragma unroll
  for (int kk = 0; kk < 4; ++kk)
    qf[kk] = cat8(*(const v8bf*)(qrow + kk * 32 + half * 16),
                  *(const v8bf*)(qrow + kk * 32 + half * 16 + 8));

  float mx[8], sm[8];
#pragma unroll
  for (int j = 0; j < 8; ++j) { mx[j] = -INFINITY; sm[j] = 0.0f; }
  v8f o[8];
#pragma unroll
  for (int n = 0; n < 8; ++n) o[n] = vzero8();

  const int nkb = (q0 >> 5) + 1;
  for (int kbi = 0; kbi < nkb; ++kbi) {
    const int kb0 = kbi * 32;
    v8f s0 = vzero8(), s1 = vzero8();
#pragma unroll
    for (int kk = 0; kk < 4; ++kk) {
      v16bf kf0 = *(const v16bf*)(kb + (bh * SS + kb0 + m) * HDIM + kk * 32 + half * 16);
      v16bf kf1 = *(const v16bf*)(kb + (bh * SS + kb0 + 16 + m) * HDIM + kk * 32 + half * 16);
      s0 = wmma_bf16(qf[kk], kf0, s0);
      s1 = wmma_bf16(qf[kk], kf1, s1);
    }
    // causal mask (only diagonal block needs it)
    if (kbi == nkb - 1) {
      int c0 = kb0 + m, c1 = c0 + 16;
#pragma unroll
      for (int j = 0; j < 8; ++j) {
        int row = q0 + j + 8 * half;
        s0[j] = (c0 <= row) ? s0[j] : -INFINITY;
        s1[j] = (c1 <= row) ? s1[j] : -INFINITY;
      }
    }
    float corr[8];
#pragma unroll
    for (int j = 0; j < 8; ++j) {
      float rm = fmaxf(s0[j], s1[j]);
#pragma unroll
      for (int d = 1; d < 16; d <<= 1) rm = fmaxf(rm, __shfl_xor(rm, d, 32));
      float mnew = fmaxf(mx[j], rm);
      corr[j] = __expf(mx[j] - mnew);
      float p0 = __expf(s0[j] - mnew);
      float p1 = __expf(s1[j] - mnew);
      float rs = p0 + p1;
#pragma unroll
      for (int d = 1; d < 16; d <<= 1) rs += __shfl_xor(rs, d, 32);
      sm[j] = sm[j] * corr[j] + rs;
      mx[j] = mnew;
      s0[j] = p0;
      s1[j] = p1;
    }
#pragma unroll
    for (int n = 0; n < 8; ++n)
#pragma unroll
      for (int j = 0; j < 8; ++j) o[n][j] *= corr[j];

    // C-layout P -> LDS -> A-layout fragment
#pragma unroll
    for (int j = 0; j < 8; ++j) {
      int row = j + 8 * half;
      pls[row * 32 + m]      = (bf16_t)s0[j];
      pls[row * 32 + 16 + m] = (bf16_t)s1[j];
    }
    __syncthreads();
    const int koff = half * 16;
    v16bf pf = cat8(*(const v8bf*)&pls[m * 32 + koff],
                    *(const v8bf*)&pls[m * 32 + koff + 8]);
    __syncthreads();

#pragma unroll
    for (int n = 0; n < 8; ++n) {
      int hd = n * 16 + m;
      v16bf vf = *(const v16bf*)(vt + (bh * HDIM + hd) * SS + kb0 + half * 16);
      o[n] = wmma_bf16(pf, vf, o[n]);
    }
  }

  float inv[8];
#pragma unroll
  for (int j = 0; j < 8; ++j) inv[j] = 1.0f / sm[j];
#pragma unroll
  for (int n = 0; n < 8; ++n)
#pragma unroll
    for (int j = 0; j < 8; ++j) {
      int row = q0 + j + 8 * half;
      int col = h * HDIM + n * 16 + m;
      ctx[(size_t)(b * SS + row) * DD + col] = (bf16_t)(o[n][j] * inv[j]);
    }
}

// ---------------------------------------------------------------- silu(a)*b -> bf16
__global__ void silu_mul(const bf16_t* __restrict__ s1, const bf16_t* __restrict__ s2,
                         bf16_t* __restrict__ out, int n) {
  int i = (blockIdx.x * 256 + threadIdx.x) * 4;
  if (i + 3 < n) {
#pragma unroll
    for (int r = 0; r < 4; ++r) {
      float a = (float)s1[i + r];
      float b = (float)s2[i + r];
      out[i + r] = (bf16_t)(a / (1.0f + __expf(-a)) * b);
    }
  }
}

// ---------------------------------------------------------------- host
extern "C" void kernel_launch(void* const* d_in, const int* in_sizes, int n_in,
                              void* d_out, int out_size, void* d_ws, size_t ws_size,
                              hipStream_t stream) {
  const float* x  = (const float*)d_in[0];
  const float* Wq = (const float*)d_in[1];
  const float* Wk = (const float*)d_in[2];
  const float* Wv = (const float*)d_in[3];
  const float* Wo = (const float*)d_in[4];
  const float* W1 = (const float*)d_in[5];
  const float* W2 = (const float*)d_in[6];
  const float* W3 = (const float*)d_in[7];
  const float* g1 = (const float*)d_in[8];
  const float* g2 = (const float*)d_in[9];
  float* out = (float*)d_out;

  char* ws = (char*)d_ws;
  size_t off = 0;
  auto alloc = [&](size_t bytes) -> void* {
    void* p = ws + off;
    off += (bytes + 255) & ~(size_t)255;
    return p;
  };
  const size_t M = (size_t)BB * SS;  // 4096
  bf16_t* Wqkvb = (bf16_t*)alloc((size_t)3 * DD * DD * 2);
  bf16_t* Wob   = (bf16_t*)alloc((size_t)DD * DD * 2);
  bf16_t* W1b   = (bf16_t*)alloc((size_t)FFD * DD * 2);
  bf16_t* W2b   = (bf16_t*)alloc((size_t)FFD * DD * 2);
  bf16_t* W3b   = (bf16_t*)alloc((size_t)DD * FFD * 2);
  bf16_t* hb    = (bf16_t*)alloc(M * DD * 2);
  float*  qkvf  = (float*)alloc(M * 3 * DD * 4);
  bf16_t* qbuf  = (bf16_t*)alloc(M * DD * 2);
  bf16_t* kbuf  = (bf16_t*)alloc(M * DD * 2);
  bf16_t* vbuf  = (bf16_t*)alloc(M * DD * 2);
  bf16_t* ctxb  = (bf16_t*)alloc(M * DD * 2);
  float*  x2    = (float*)alloc(M * DD * 4);
  bf16_t* h2b   = (bf16_t*)alloc(M * DD * 2);
  bf16_t* s1b   = (bf16_t*)alloc(M * FFD * 2);
  bf16_t* s2b   = (bf16_t*)alloc(M * FFD * 2);
  bf16_t* ffhb  = (bf16_t*)alloc(M * FFD * 2);

  // weights -> bf16 (Wq/Wk/Wv concatenated into one [3D, D] matrix)
  int nDD = DD * DD, nFD = FFD * DD;
  cvt_f32_bf16<<<nDD / 1024, 256, 0, stream>>>(Wq, Wqkvb, nDD);
  cvt_f32_bf16<<<nDD / 1024, 256, 0, stream>>>(Wk, Wqkvb + (size_t)nDD, nDD);
  cvt_f32_bf16<<<nDD / 1024, 256, 0, stream>>>(Wv, Wqkvb + (size_t)2 * nDD, nDD);
  cvt_f32_bf16<<<nDD / 1024, 256, 0, stream>>>(Wo, Wob, nDD);
  cvt_f32_bf16<<<nFD / 1024, 256, 0, stream>>>(W1, W1b, nFD);
  cvt_f32_bf16<<<nFD / 1024, 256, 0, stream>>>(W2, W2b, nFD);
  cvt_f32_bf16<<<nFD / 1024, 256, 0, stream>>>(W3, W3b, nFD);

  // h = rmsnorm(x, g1)
  rmsnorm_bf16<<<(int)M, 256, 0, stream>>>(x, g1, hb);
  // qkv = h @ Wqkv^T   [4096, 6144] f32
  gemm_bf16<0><<<dim3(3 * DD / 128, (int)M / 128), 256, 0, stream>>>(
      hb, Wqkvb, qkvf, nullptr, (int)M, 3 * DD, DD);
  // rope + split + V transpose
  rope_split<<<(BB * SS * HH * 64) / 256, 256, 0, stream>>>(qkvf, qbuf, kbuf, vbuf);
  // attention
  attn_kernel<<<BB * HH * (SS / 16), 32, 0, stream>>>(qbuf, kbuf, vbuf, ctxb);
  // x2 = x + ctx @ Wo^T
  gemm_bf16<2><<<dim3(DD / 128, (int)M / 128), 256, 0, stream>>>(
      ctxb, Wob, x2, x, (int)M, DD, DD);
  // h2 = rmsnorm(x2, g2)
  rmsnorm_bf16<<<(int)M, 256, 0, stream>>>(x2, g2, h2b);
  // s1 = h2 @ W1^T ; s2 = h2 @ W2^T (bf16)
  gemm_bf16<1><<<dim3(FFD / 128, (int)M / 128), 256, 0, stream>>>(
      h2b, W1b, s1b, nullptr, (int)M, FFD, DD);
  gemm_bf16<1><<<dim3(FFD / 128, (int)M / 128), 256, 0, stream>>>(
      h2b, W2b, s2b, nullptr, (int)M, FFD, DD);
  // ffh = silu(s1) * s2
  silu_mul<<<(int)(M * FFD / 1024), 256, 0, stream>>>(s1b, s2b, ffhb, (int)(M * FFD));
  // out = x2 + ffh @ W3^T
  gemm_bf16<2><<<dim3(DD / 128, (int)M / 128), 256, 0, stream>>>(
      ffhb, W3b, out, x2, (int)M, DD, FFD);
}
